// PicoGPT_69758858821805
// MI455X (gfx1250) — matrix operand order repaired and checked by reading, not compiled, not run on hardware
//
#include <hip/hip_runtime.h>
#include <hip/hip_bf16.h>
#include <math.h>

// Model dims (PicoGPT reference)
#define N_B   2
#define T_S   1024
#define D_M   1024
#define H_N   16
#define D_K   64
#define D_H   4096
#define L_N   8
#define V_S   32000
#define M_ROWS (N_B * T_S)   // 2048

typedef __attribute__((ext_vector_type(16))) __bf16 v16bf;
typedef __attribute__((ext_vector_type(8)))  float  v8f;

union FragBF { uint4 u[2]; v16bf v; };

__device__ __forceinline__ unsigned short f2bf(float f) {
  unsigned u = __float_as_uint(f);
  u += 0x7FFFu + ((u >> 16) & 1u);     // round-to-nearest-even
  return (unsigned short)(u >> 16);
}
__device__ __forceinline__ void st4bf(unsigned short* dst, float4 v) {
  ushort4 s;
  s.x = f2bf(v.x); s.y = f2bf(v.y); s.z = f2bf(v.z); s.w = f2bf(v.w);
  *reinterpret_cast<ushort4*>(dst) = s;
}
__device__ __forceinline__ float gelu_f(float x) {
  float x3 = x * x * x;
  return 0.5f * x * (1.0f + tanhf(0.7978845608028654f * (x + 0.044715f * x3)));
}

// ---------------------------------------------------------------------------
// Generic NT GEMM: C[M,N] = act( scale * A[M,K] x W[N,K]^T + bias ) (+C)
// Block tile 128(M) x 128(N) x 32(K); 8 waves in 4x2 grid, each wave 32x64
// (2 A-frags x 4 B-frags = 8 WMMA per K-step). Double-buffered LDS so the
// next K-slab's global loads overlap the current slab's WMMAs.
// Batched via blockIdx.z with (outer,inner) offset pairs: off = z/Hdiv*out + z%Hdiv*in.
// ---------------------------------------------------------------------------
template<int ACT, int MASK, int RESID>
__global__ __launch_bounds__(256) void gemm_nt(
    const float* __restrict__ A, const float* __restrict__ B,
    const float* __restrict__ bias, float* __restrict__ C,
    int K, int lda, int ldb, int ldc, float scale, int Hdiv,
    long long aOut, long long aIn, long long bOut, long long bIn,
    long long cOut, long long cIn)
{
  __shared__ __align__(16) unsigned short sA[2][128 * 40];
  __shared__ __align__(16) unsigned short sB[2][128 * 40];

  const int tid  = threadIdx.x;
  const int lane = tid & 31, wave = tid >> 5;
  const int wm   = wave & 3, wn = wave >> 2;     // 4 x 2 wave grid
  const int bm   = blockIdx.y, bn = blockIdx.x;
  const int z    = blockIdx.z;

  A += (long long)(z / Hdiv) * aOut + (long long)(z % Hdiv) * aIn;
  B += (long long)(z / Hdiv) * bOut + (long long)(z % Hdiv) * bIn;
  C += (long long)(z / Hdiv) * cOut + (long long)(z % Hdiv) * cIn;

  const float* Ab = A + (long long)(bm * 128) * lda;
  const float* Bb = B + (long long)(bn * 128) * ldb;

  v8f acc[2][4];
#pragma unroll
  for (int i = 0; i < 2; i++)
#pragma unroll
    for (int j = 0; j < 4; j++)
#pragma unroll
      for (int e = 0; e < 8; e++) acc[i][j][e] = 0.0f;

  const int hi8   = (lane >> 4) * 8;
  const int hi16  = (lane >> 4) * 16;
  const int arow0 = wm * 32 + (lane & 15);
  const int arow1 = arow0 + 16;

  auto ldglob = [&](int k0, float4* ra, float4* rb) {
#pragma unroll
    for (int j = 0; j < 4; j++) {
      const int f = tid + j * 256, r = f >> 3, c = (f & 7) << 2;
      ra[j] = *reinterpret_cast<const float4*>(Ab + (long long)r * lda + k0 + c);
      rb[j] = *reinterpret_cast<const float4*>(Bb + (long long)r * ldb + k0 + c);
    }
  };
  auto st_lds = [&](int buf, const float4* ra, const float4* rb) {
#pragma unroll
    for (int j = 0; j < 4; j++) {
      const int f = tid + j * 256, r = f >> 3, c = (f & 7) << 2;
      st4bf(&sA[buf][r * 40 + c], ra[j]);
      st4bf(&sB[buf][r * 40 + c], rb[j]);
    }
  };
  auto compute = [&](int buf) {
    FragBF a0, a1;
    a0.u[0] = *reinterpret_cast<const uint4*>(&sA[buf][arow0 * 40 + hi8]);
    a0.u[1] = *reinterpret_cast<const uint4*>(&sA[buf][arow0 * 40 + 16 + hi8]);
    a1.u[0] = *reinterpret_cast<const uint4*>(&sA[buf][arow1 * 40 + hi8]);
    a1.u[1] = *reinterpret_cast<const uint4*>(&sA[buf][arow1 * 40 + 16 + hi8]);
#pragma unroll
    for (int nt = 0; nt < 4; nt++) {
      FragBF b;
      const int boff = (wn * 64 + nt * 16 + (lane & 15)) * 40 + hi8;
      b.u[0] = *reinterpret_cast<const uint4*>(&sB[buf][boff]);
      b.u[1] = *reinterpret_cast<const uint4*>(&sB[buf][boff + 16]);
      acc[0][nt] = __builtin_amdgcn_wmma_f32_16x16x32_bf16(
          false, a0.v, false, b.v, (short)0, acc[0][nt], false, false);
      acc[1][nt] = __builtin_amdgcn_wmma_f32_16x16x32_bf16(
          false, a1.v, false, b.v, (short)0, acc[1][nt], false, false);
    }
  };

  {                                               // prologue: slab 0 -> buf 0
    float4 ra[4], rb[4];
    ldglob(0, ra, rb);
    st_lds(0, ra, rb);
  }
  __syncthreads();

  const int nsteps = K >> 5;
  int cur = 0;
  for (int s = 1; s < nsteps; s++) {
    float4 ra[4], rb[4];
    ldglob(s * 32, ra, rb);                       // in flight during compute
    if (s + 1 < nsteps)                           // prefetch slab s+1 (weights)
      __builtin_prefetch(Bb + (long long)(tid >> 1) * ldb + s * 32 + 32, 0, 0);
    compute(cur);
    st_lds(cur ^ 1, ra, rb);
    __syncthreads();
    cur ^= 1;
  }
  compute(cur);

  const int mbase = bm * 128 + wm * 32 + 8 * (lane >> 4);
  const int nbase = bn * 128 + wn * 64 + (lane & 15);
#pragma unroll
  for (int nt = 0; nt < 4; nt++) {
    const int n = nbase + nt * 16;
    const float bv = bias ? bias[n] : 0.0f;
#pragma unroll
    for (int mi = 0; mi < 2; mi++) {
#pragma unroll
      for (int r = 0; r < 8; r++) {
        const int mrow = mbase + mi * 16 + r;
        float val = acc[mi][nt][r] * scale + bv;
        if (ACT)  val = gelu_f(val);
        if (MASK) { if (n > mrow) val = -1e30f; }  // causal: key s > query t
        float* cp = C + (long long)mrow * ldc + n;
        if (RESID) val += *cp;
        *cp = val;
      }
    }
  }
}

// ---------------------------------------------------------------------------
// NN GEMM with accumulate: C[M,N] += A[M,K] x B[K,N]   (attention P·V)
// Block tile 128(M) x 64(N) x 32(K); 8 waves stacked along M, each 16x64.
// B tile is transposed during the LDS store so fragments read contiguously.
// ---------------------------------------------------------------------------
__global__ __launch_bounds__(256) void gemm_nn_acc(
    const float* __restrict__ A, const float* __restrict__ B, float* __restrict__ C,
    int K, int lda, int ldb, int ldc, int Hdiv,
    long long aOut, long long aIn, long long bOut, long long bIn,
    long long cOut, long long cIn)
{
  __shared__ __align__(16) unsigned short sA[128 * 40];
  __shared__ __align__(16) unsigned short sB[64 * 40];   // [n][k] (transposed)

  const int tid  = threadIdx.x;
  const int lane = tid & 31, wave = tid >> 5;
  const int bm   = blockIdx.y;
  const int z    = blockIdx.z;

  A += (long long)(z / Hdiv) * aOut + (long long)(z % Hdiv) * aIn;
  B += (long long)(z / Hdiv) * bOut + (long long)(z % Hdiv) * bIn;
  C += (long long)(z / Hdiv) * cOut + (long long)(z % Hdiv) * cIn;

  v8f acc[4];
#pragma unroll
  for (int i = 0; i < 4; i++)
#pragma unroll
    for (int j = 0; j < 8; j++) acc[i][j] = 0.0f;

  const int hi8  = (lane >> 4) * 8;
  const int hi16 = (lane >> 4) * 16;
  const int arow = wave * 16 + (lane & 15);

  for (int k0 = 0; k0 < K; k0 += 32) {
#pragma unroll
    for (int j = 0; j < 4; j++) {                 // A tile 128x32
      int f = tid + j * 256, r = f >> 3, c = (f & 7) << 2;
      st4bf(&sA[r * 40 + c],
            *reinterpret_cast<const float4*>(A + (long long)(bm * 128 + r) * lda + k0 + c));
    }
#pragma unroll
    for (int j = 0; j < 2; j++) {                 // B tile 32x64, store transposed
      int f = tid + j * 256, kk = f >> 4, c = (f & 15) << 2;
      float4 v = *reinterpret_cast<const float4*>(B + (long long)(k0 + kk) * ldb + c);
      sB[(c + 0) * 40 + kk] = f2bf(v.x);
      sB[(c + 1) * 40 + kk] = f2bf(v.y);
      sB[(c + 2) * 40 + kk] = f2bf(v.z);
      sB[(c + 3) * 40 + kk] = f2bf(v.w);
    }
    __syncthreads();

    FragBF a;
    a.u[0] = *reinterpret_cast<const uint4*>(&sA[arow * 40 + hi8]);
    a.u[1] = *reinterpret_cast<const uint4*>(&sA[arow * 40 + 16 + hi8]);
#pragma unroll
    for (int nt = 0; nt < 4; nt++) {
      FragBF b;
      const int boff = (nt * 16 + (lane & 15)) * 40 + hi8;
      b.u[0] = *reinterpret_cast<const uint4*>(&sB[boff]);
      b.u[1] = *reinterpret_cast<const uint4*>(&sB[boff + 16]);
      acc[nt] = __builtin_amdgcn_wmma_f32_16x16x32_bf16(
          false, a.v, false, b.v, (short)0, acc[nt], false, false);
    }
    __syncthreads();
  }

  const int mloc = bm * 128 + wave * 16 + 8 * (lane >> 4);
#pragma unroll
  for (int nt = 0; nt < 4; nt++) {
    const int n = nt * 16 + (lane & 15);
#pragma unroll
    for (int r = 0; r < 8; r++) {
      float* cp = C + (long long)(mloc + r) * ldc + n;
      *cp += acc[nt][r];
    }
  }
}

// ---------------------------------------------------------------------------
// Embedding + sinusoidal positional encoding: x[nt][d] = emb[tok[nt]][d] + pe[t][d]
// ---------------------------------------------------------------------------
__global__ __launch_bounds__(256) void embed_pe(const int* __restrict__ tok,
                                                const float* __restrict__ emb,
                                                float* __restrict__ x)
{
  const long long idx = (long long)blockIdx.x * 256 + threadIdx.x;  // < 2048*1024
  const int d  = (int)(idx & (D_M - 1));
  const int nt = (int)(idx >> 10);
  const int t  = nt & (T_S - 1);
  const float div = __expf((float)(d & ~1) * (-9.210340371976184f / (float)D_M));
  const float ang = (float)t * div;
  const float pe  = (d & 1) ? __cosf(ang) : __sinf(ang);
  x[idx] = emb[(long long)tok[nt] * D_M + d] + pe;
}

// ---------------------------------------------------------------------------
// LayerNorm over last dim (1024); one 256-thread block per row, 4 elems/thread.
// ---------------------------------------------------------------------------
__global__ __launch_bounds__(256) void layernorm_k(const float* __restrict__ x,
                                                   const float* __restrict__ w,
                                                   const float* __restrict__ b,
                                                   float* __restrict__ o)
{
  __shared__ float r1[8];
  __shared__ float r2[8];
  const int tid = threadIdx.x;
  const long long row = blockIdx.x;
  const float4 v = reinterpret_cast<const float4*>(x + row * D_M)[tid];

  float s = v.x + v.y + v.z + v.w;
#pragma unroll
  for (int o_ = 16; o_ > 0; o_ >>= 1) s += __shfl_down(s, o_, 32);
  if ((tid & 31) == 0) r1[tid >> 5] = s;
  __syncthreads();
  float mu = 0.0f;
#pragma unroll
  for (int i = 0; i < 8; i++) mu += r1[i];
  mu *= (1.0f / (float)D_M);

  const float dx = v.x - mu, dy = v.y - mu, dz = v.z - mu, dw = v.w - mu;
  float s2 = dx * dx + dy * dy + dz * dz + dw * dw;
#pragma unroll
  for (int o_ = 16; o_ > 0; o_ >>= 1) s2 += __shfl_down(s2, o_, 32);
  if ((tid & 31) == 0) r2[tid >> 5] = s2;
  __syncthreads();
  float var = 0.0f;
#pragma unroll
  for (int i = 0; i < 8; i++) var += r2[i];
  var *= (1.0f / (float)D_M);
  const float inv = rsqrtf(var + 1e-5f);

  const float4 wv = reinterpret_cast<const float4*>(w)[tid];
  const float4 bv = reinterpret_cast<const float4*>(b)[tid];
  float4 out;
  out.x = dx * inv * wv.x + bv.x;
  out.y = dy * inv * wv.y + bv.y;
  out.z = dz * inv * wv.z + bv.z;
  out.w = dw * inv * wv.w + bv.w;
  reinterpret_cast<float4*>(o + row * D_M)[tid] = out;
}

// ---------------------------------------------------------------------------
// In-place softmax over rows of length T_S (masked entries hold -1e30 -> exp=0).
// ---------------------------------------------------------------------------
__global__ __launch_bounds__(256) void softmax_k(float* __restrict__ S)
{
  __shared__ float r1[8];
  __shared__ float r2[8];
  const int tid = threadIdx.x;
  float4* p = reinterpret_cast<float4*>(S + (long long)blockIdx.x * T_S);
  float4 v = p[tid];

  float mx = fmaxf(fmaxf(v.x, v.y), fmaxf(v.z, v.w));
#pragma unroll
  for (int o_ = 16; o_ > 0; o_ >>= 1) mx = fmaxf(mx, __shfl_down(mx, o_, 32));
  if ((tid & 31) == 0) r1[tid >> 5] = mx;
  __syncthreads();
  float bm = -1e30f;
#pragma unroll
  for (int i = 0; i < 8; i++) bm = fmaxf(bm, r1[i]);

  float4 e;
  e.x = __expf(v.x - bm); e.y = __expf(v.y - bm);
  e.z = __expf(v.z - bm); e.w = __expf(v.w - bm);
  float s = e.x + e.y + e.z + e.w;
#pragma unroll
  for (int o_ = 16; o_ > 0; o_ >>= 1) s += __shfl_down(s, o_, 32);
  if ((tid & 31) == 0) r2[tid >> 5] = s;
  __syncthreads();
  float tot = 0.0f;
#pragma unroll
  for (int i = 0; i < 8; i++) tot += r2[i];
  const float inv = 1.0f / tot;
  e.x *= inv; e.y *= inv; e.z *= inv; e.w *= inv;
  p[tid] = e;
}

// ---------------------------------------------------------------------------
// Host-side orchestration (graph-capture safe: all launches on `stream`).
// Workspace layout (bytes): x 0..8M, h 8..16M, q 16..24M, k 24..32M,
// v 32..40M, mlp 40..72M, S 72..200M.
// ---------------------------------------------------------------------------
extern "C" void kernel_launch(void* const* d_in, const int* in_sizes, int n_in,
                              void* d_out, int out_size, void* d_ws, size_t ws_size,
                              hipStream_t stream)
{
  (void)in_sizes; (void)n_in; (void)out_size; (void)ws_size;

  const int*   tok  = (const int*)  d_in[0];
  const float* emb  = (const float*)d_in[1];
  const float* ln1w = (const float*)d_in[2];
  const float* ln1b = (const float*)d_in[3];
  const float* wq   = (const float*)d_in[4];
  const float* wk   = (const float*)d_in[5];
  const float* wv   = (const float*)d_in[6];
  const float* ln2w = (const float*)d_in[7];
  const float* ln2b = (const float*)d_in[8];
  const float* fc1w = (const float*)d_in[9];
  const float* fc1b = (const float*)d_in[10];
  const float* fc2w = (const float*)d_in[11];
  const float* fc2b = (const float*)d_in[12];
  const float* roW  = (const float*)d_in[13];
  const float* roB  = (const float*)d_in[14];
  float* out = (float*)d_out;

  char* ws = (char*)d_ws;
  float* x = (float*)(ws);
  float* h = (float*)(ws + (8ll  << 20));
  float* q = (float*)(ws + (16ll << 20));
  float* k = (float*)(ws + (24ll << 20));
  float* v = (float*)(ws + (32ll << 20));
  float* m = (float*)(ws + (40ll << 20));
  float* S = (float*)(ws + (72ll << 20));

  const dim3 blk(256);
  const long long TT = (long long)T_S * T_S;

  embed_pe<<<dim3((M_ROWS * D_M) / 256), blk, 0, stream>>>(tok, emb, x);

  for (int l = 0; l < L_N; l++) {
    // --- attention block ---
    layernorm_k<<<dim3(M_ROWS), blk, 0, stream>>>(x, ln1w + (long long)l * D_M,
                                                  ln1b + (long long)l * D_M, h);
    const long long wqkv_off = (long long)l * H_N * D_K * D_M;
    gemm_nt<0, 0, 0><<<dim3(D_M / 128, M_ROWS / 128, 1), blk, 0, stream>>>(
        h, wq + wqkv_off, nullptr, q, D_M, D_M, D_M, D_M, 1.0f, 1, 0, 0, 0, 0, 0, 0);
    gemm_nt<0, 0, 0><<<dim3(D_M / 128, M_ROWS / 128, 1), blk, 0, stream>>>(
        h, wk + wqkv_off, nullptr, k, D_M, D_M, D_M, D_M, 1.0f, 1, 0, 0, 0, 0, 0, 0);
    gemm_nt<0, 0, 0><<<dim3(D_M / 128, M_ROWS / 128, 1), blk, 0, stream>>>(
        h, wv + wqkv_off, nullptr, v, D_M, D_M, D_M, D_M, 1.0f, 1, 0, 0, 0, 0, 0, 0);

    // S = 0.125 * Q K^T with causal mask, batched over z = n*H + h_idx
    gemm_nt<0, 1, 0><<<dim3(T_S / 128, T_S / 128, N_B * H_N), blk, 0, stream>>>(
        q, k, nullptr, S, D_K, D_M, D_M, T_S, 0.125f, H_N,
        (long long)T_S * D_M, 64, (long long)T_S * D_M, 64,
        (long long)H_N * TT, TT);

    softmax_k<<<dim3(N_B * H_N * T_S), blk, 0, stream>>>(S);

    // x += P V  (accumulate into residual stream, columns h_idx*64..+63)
    gemm_nn_acc<<<dim3(1, T_S / 128, N_B * H_N), blk, 0, stream>>>(
        S, v, x, T_S, T_S, D_M, D_M, H_N,
        (long long)H_N * TT, TT,
        (long long)T_S * D_M, 64,
        (long long)T_S * D_M, 64);

    // --- MLP block ---
    layernorm_k<<<dim3(M_ROWS), blk, 0, stream>>>(x, ln2w + (long long)l * D_M,
                                                  ln2b + (long long)l * D_M, h);
    gemm_nt<1, 0, 0><<<dim3(D_H / 128, M_ROWS / 128, 1), blk, 0, stream>>>(
        h, fc1w + (long long)l * D_H * D_M, fc1b + (long long)l * D_H, m,
        D_M, D_M, D_M, D_H, 1.0f, 1, 0, 0, 0, 0, 0, 0);
    gemm_nt<0, 0, 1><<<dim3(D_M / 128, M_ROWS / 128, 1), blk, 0, stream>>>(
        m, fc2w + (long long)l * D_M * D_H, fc2b + (long long)l * D_M, x,
        D_H, D_H, D_H, D_M, 1.0f, 1, 0, 0, 0, 0, 0, 0);
  }

  // readout: out[2048][32000] = x · ro_w^T + ro_b
  gemm_nt<0, 0, 0><<<dim3(V_S / 128, M_ROWS / 128, 1), blk, 0, stream>>>(
      x, roW, roB, out, D_M, D_M, D_M, V_S, 1.0f, 1, 0, 0, 0, 0, 0, 0);
}
